// MultiHeadAttention_3496103379159
// MI455X (gfx1250) — matrix operand order repaired and checked by reading, not compiled, not run on hardware
//
#include <hip/hip_runtime.h>
#include <hip/hip_bf16.h>
#include <hip/hip_fp16.h>

typedef __attribute__((ext_vector_type(16))) _Float16 v16h;
typedef __attribute__((ext_vector_type(8)))  _Float16 v8h;
typedef __attribute__((ext_vector_type(8)))  float    v8f;
typedef __attribute__((ext_vector_type(4)))  float    v4f;
typedef __attribute__((ext_vector_type(4)))  unsigned int v4u;
typedef __attribute__((ext_vector_type(8)))  int      v8i;
typedef __attribute__((ext_vector_type(4)))  int      v4i;

#define BB 4
#define SS 2048
#define DD 1024
#define NH 16
#define DH 64

static __device__ __forceinline__ v8f wmma_f16(v16h a, v16h b, v8f c) {
  // D = A(16x32 f16) * B(32x16 f16) + C(16x16 f32)
  return __builtin_amdgcn_wmma_f32_16x16x32_f16(false, a, false, b, (short)0, c,
                                                false, false);
}

static __device__ __forceinline__ v16h cat16(v8h lo, v8h hi) {
  v16h r;
#pragma unroll
  for (int i = 0; i < 8; ++i) { r[i] = lo[i]; r[i + 8] = hi[i]; }
  return r;
}

// A fragment (16x32, f16): lane l holds row m=l&15; lanes 0-15 K={0..7,16..23},
// lanes 16-31 K={8..15,24..31}.  p points at element [0][0], row stride ld halves.
static __device__ __forceinline__ v16h loadA(const _Float16* p, int ld, int lane) {
  const int m  = lane & 15;
  const int k0 = (lane >> 4) << 3;          // 0 or 8
  const _Float16* r = p + (size_t)m * ld + k0;
  v8h lo = *(const v8h*)(r);
  v8h hi = *(const v8h*)(r + 16);
  return cat16(lo, hi);
}

// B fragment (32x16, f16), source stored column-major-as-rows: BT[n][k].
// lane l holds column n=l&15; lanes 0-15 K={0..7,16..23}, lanes 16-31 K={8..15,24..31}.
static __device__ __forceinline__ v16h loadB(const _Float16* p, int ld, int lane) {
  const int n  = lane & 15;
  const int k0 = (lane >> 4) << 3;          // 0 or 8
  const _Float16* r = p + (size_t)n * ld + k0;
  v8h lo = *(const v8h*)(r);
  v8h hi = *(const v8h*)(r + 16);
  return cat16(lo, hi);
}

// TDM: DMA a 2D f16 tile (d0 contiguous elems per row, d1 rows, row stride
// `stride0` elems) global -> LDS, packed [d1][d0].  Tiles are always fully
// in-bounds so tensor dims = tile dims.  Tracked on TENSORcnt.
static __device__ __forceinline__ void tdm_load_2d(const _Float16* gsrc,
                                                   _Float16* lds_dst,
                                                   unsigned d0, unsigned d1,
                                                   unsigned stride0) {
  const unsigned long long ga = (unsigned long long)(uintptr_t)gsrc;
  const unsigned ldsa = (unsigned)(uintptr_t)lds_dst;   // low 32 bits = LDS offset
  v4u g0;
  g0[0] = 1u;                                           // count=1, user descriptor
  g0[1] = ldsa;                                         // lds_addr [63:32]
  g0[2] = (unsigned)(ga & 0xFFFFFFFFu);                 // global_addr lo
  g0[3] = (unsigned)((ga >> 32) & 0x01FFFFFFu) | (2u << 30);  // addr hi | type=2
  v8i g1;
  g1[0] = 0x00010000;                    // workgroup_mask=0, data_size=1 (2B)
  g1[1] = (int)(d0 << 16);               // tensor_dim0 lo16 (bits 63:48)
  g1[2] = (int)(d1 << 16);               // tensor_dim0 hi | tensor_dim1 lo16
  g1[3] = (int)(d0 << 16);               // tensor_dim1 hi | tile_dim0
  g1[4] = (int)d1;                       // tile_dim1 | tile_dim2=0
  g1[5] = (int)stride0;                  // tensor_dim0_stride lo32
  g1[6] = 0;
  g1[7] = 0;
  v4i g2 = {0, 0, 0, 0};                 // 2D tensor: dims 2+ unused
  v4i g3 = {0, 0, 0, 0};
#if defined(__clang_major__) && __clang_major__ >= 23
  v8i g4 = {0, 0, 0, 0, 0, 0, 0, 0};
  __builtin_amdgcn_tensor_load_to_lds(g0, g1, g2, g3, g4, 0);
#else
  __builtin_amdgcn_tensor_load_to_lds(g0, g1, g2, g3, 0);
#endif
}

// ---------------------------------------------------------------- converters
__global__ __launch_bounds__(256) void f32_to_f16_kernel(
    const float* __restrict__ src, _Float16* __restrict__ dst, int n) {
  int i = blockIdx.x * 256 + threadIdx.x;
  if (i < n) dst[i] = (_Float16)src[i];
}

// wqkvT: [3072][1024] = transpose of [Wq|Wk|Wv]; woT: [1024][1024] = Wo^T.
__global__ __launch_bounds__(256) void prep_weights_kernel(
    const float* __restrict__ Wq, const float* __restrict__ Wk,
    const float* __restrict__ Wv, const float* __restrict__ Wo,
    _Float16* __restrict__ wqkvT, _Float16* __restrict__ woT) {
  int i = blockIdx.x * 256 + threadIdx.x;
  if (i < 3 * 1024 * 1024) {
    int j = i >> 10, kk = i & 1023;
    int which = j >> 10, n = j & 1023;
    const float* W = (which == 0) ? Wq : (which == 1 ? Wk : Wv);
    wqkvT[i] = (_Float16)W[kk * 1024 + n];
  } else if (i < 4 * 1024 * 1024) {
    int t = i - 3 * 1024 * 1024;
    int n = t >> 10, kk = t & 1023;
    woT[t] = (_Float16)Wo[kk * 1024 + n];
  }
}

// ---------------------------------------------------------------- GEMM
// Block = 128 (4 waves), workgroup tile 64(M) x 128(N).  The 64x32 A tile is
// DMA'd into LDS by the TDM (double-buffered, wave 0) and shared by all waves;
// B fragments stream from L2.  8 WMMA per k-step.
// MODE 0: write f32 + bias (out projection).  MODE 1: scatter f16 Q/K/V^T.
template <int MODE>
__global__ __launch_bounds__(128) void gemm_kernel(
    const _Float16* __restrict__ A, const _Float16* __restrict__ BT,
    const float* __restrict__ bias0, const float* __restrict__ bias1,
    const float* __restrict__ bias2, float* __restrict__ outF,
    _Float16* __restrict__ qb, _Float16* __restrict__ kb,
    _Float16* __restrict__ vtb) {
  __shared__ __align__(16) _Float16 astage[2][64 * 32];  // 2 x 4 KB A tiles
  const int lane = threadIdx.x & 31;
  const int wave = __builtin_amdgcn_readfirstlane((int)(threadIdx.x >> 5));
  const int n0 = blockIdx.x * 128 + wave * 16;     // second subtile at n0+64
  const int m0 = blockIdx.y * 64;

  const _Float16* Arow  = A + (size_t)m0 * DD;
  const _Float16* Brow0 = BT + (size_t)n0 * DD;
  const _Float16* Brow1 = BT + (size_t)(n0 + 64) * DD;

  if (wave == 0) {
    tdm_load_2d(Arow, &astage[0][0], 32, 64, DD);
    __builtin_amdgcn_s_wait_tensorcnt(0);
  }
  __syncthreads();

  v8f acc[4][2] = {};
  for (int ks = 0; ks < DD / 32; ++ks) {
    const int cur = ks & 1;
    if (wave == 0 && ks + 1 < DD / 32)
      tdm_load_2d(Arow + (ks + 1) * 32, &astage[cur ^ 1][0], 32, 64, DD);
    v16h bf0 = loadB(Brow0 + ks * 32, DD, lane);
    v16h bf1 = loadB(Brow1 + ks * 32, DD, lane);
#pragma unroll
    for (int t = 0; t < 4; ++t) {
      v16h af = loadA(&astage[cur][t * 16 * 32], 32, lane);
      acc[t][0] = wmma_f16(af, bf0, acc[t][0]);
      acc[t][1] = wmma_f16(af, bf1, acc[t][1]);
    }
    if (wave == 0) __builtin_amdgcn_s_wait_tensorcnt(0);
    __syncthreads();
  }

  const int nloc = lane & 15;
  const int hi = lane >> 4;
#pragma unroll
  for (int t = 0; t < 4; ++t) {
    const int mb = m0 + t * 16;          // scalar; 16-row tile stays in one seq
#pragma unroll
    for (int j = 0; j < 2; ++j) {
      const int nb = n0 + j * 64;        // scalar; 16-col tile stays in one head
#pragma unroll
      for (int r = 0; r < 8; ++r) {
        float v = acc[t][j][r];
        if (MODE == 0) {
          outF[(size_t)(mb + r + 8 * hi) * DD + nb + nloc] = v + bias0[nb + nloc];
        } else {
          const int which = nb >> 10;    // 0=Q 1=K 2=V (scalar)
          const int jb = nb & 1023;
          const int head = jb >> 6;      // scalar
          const int d = (jb & 63) + nloc;
          const int b = mb >> 11;        // scalar (S = 2048)
          const int s = (mb & 2047) + r + 8 * hi;
          const float* bias = (which == 0) ? bias0 : (which == 1 ? bias1 : bias2);
          _Float16 h16 = (_Float16)(v + bias[jb + nloc]);
          if (which == 0)
            qb[((size_t)(b * NH + head) * SS + s) * DH + d] = h16;
          else if (which == 1)
            kb[((size_t)(b * NH + head) * SS + s) * DH + d] = h16;
          else
            vtb[((size_t)(b * NH + head) * DH + d) * SS + s] = h16;  // V^T
        }
      }
    }
  }
}

// ---------------------------------------------------------------- attention
// grid = (S/64, B*NH), block = 128 (4 waves), wave w: q rows [g*64+16w, +15].
// K tiles (16x64) and V tiles (64x32) are DMA'd into LDS by the TDM
// (double-buffered; K by wave 0, V by wave 1; published via s_wait_tensorcnt +
// barrier) and shared by all waves.  Scores computed transposed (S^T = K@Q^T)
// so softmax is in-lane + one shfl_xor(16).  Pass A: online max/sum.  Pass B:
// recompute, write normalized P (b128s), ctx += P@V via WMMA (LDS P-stage).
__global__ __launch_bounds__(128) void attn_kernel(
    const _Float16* __restrict__ q16, const _Float16* __restrict__ k16,
    const _Float16* __restrict__ vT16, _Float16* __restrict__ ctx16,
    float* __restrict__ prob) {
  __shared__ __align__(16) _Float16 kstage[2][16 * DH];  // 2 x 2 KB K tiles
  __shared__ __align__(16) _Float16 vstage[2][64 * 32];  // 2 x 4 KB V tiles
  __shared__ __align__(16) _Float16 pb[4][16 * 32];      // per-wave P stage
  const int lane = threadIdx.x & 31;
  const int wave = __builtin_amdgcn_readfirstlane((int)(threadIdx.x >> 5));
  const int bh = blockIdx.y;          // b*16 + h
  const int b = bh >> 4;
  const int h = bh & 15;
  const int mbase = blockIdx.x * 64 + wave * 16;   // first q row (SGPR)
  const int ktLim = (mbase + 15) >> 4;             // wave's last causal tile
  const int ktMax = blockIdx.x * 4 + 3;            // workgroup-uniform bound
  const float scale = 0.125f;                      // 1/sqrt(64)

  const _Float16* Q  = q16 + ((size_t)bh * SS + mbase) * DH;
  const _Float16* Kp = k16 + (size_t)bh * SS * DH;
  const _Float16* VT = vT16 + (size_t)bh * DH * SS;
  float* P = prob + ((size_t)bh * SS + mbase) * SS;

  // Q rows as B-operand fragments (lane half h holds d 8h..8h+7, 16+8h..23+8h)
  const v16h qb0 = loadB(Q, DH, lane);        // d = 0..31
  const v16h qb1 = loadB(Q + 32, DH, lane);   // d = 32..63
  const int nloc = lane & 15;
  const int hi = lane >> 4;
  const int qrow = mbase + nloc;              // this lane's q row

  // ---- pass A: per-lane online softmax statistics ----
  float rmax = -3.0e38f, rsum = 0.0f;
  if (wave == 0) {
    tdm_load_2d(Kp, &kstage[0][0], 64, 16, 64);
    __builtin_amdgcn_s_wait_tensorcnt(0);
  }
  __syncthreads();
  for (int kt = 0; kt <= ktMax; ++kt) {
    const int cur = kt & 1;
    if (wave == 0 && kt < ktMax)
      tdm_load_2d(Kp + (size_t)(kt + 1) * 16 * DH, &kstage[cur ^ 1][0], 64, 16, 64);
    if (kt <= ktLim) {                        // scalar causal bound
      const _Float16* Krow = &kstage[cur][0];
      v16h ka0 = loadA(Krow, DH, lane);
      v16h ka1 = loadA(Krow + 32, DH, lane);
      v8f st = {};                 // st[r] = S^T[key = kt*16+8*hi+r][qrow]
      st = wmma_f16(ka0, qb0, st);
      st = wmma_f16(ka1, qb1, st);
      float sv[8];
#pragma unroll
      for (int r = 0; r < 8; ++r) {
        int key = kt * 16 + 8 * hi + r;
        float t = st[r] * scale;
        sv[r] = (key > qrow) ? -1.0e9f : t;
      }
      float tm = sv[0];
#pragma unroll
      for (int r = 1; r < 8; ++r) tm = fmaxf(tm, sv[r]);
      tm = fmaxf(tm, __shfl_xor(tm, 16, 32));   // other 8 keys of this column
      float nm = fmaxf(rmax, tm);
      float es = 0.0f;
#pragma unroll
      for (int r = 0; r < 8; ++r) es += __expf(sv[r] - nm);
      es += __shfl_xor(es, 16, 32);
      rsum = rsum * __expf(rmax - nm) + es;
      rmax = nm;
    }
    if (wave == 0) __builtin_amdgcn_s_wait_tensorcnt(0);
    __syncthreads();
  }
  const float rinv = 1.0f / rsum;

  // ---- pass B: write normalized P, accumulate context ----
  v8f cacc[4] = {};
  _Float16* myp = pb[wave];

  if (wave == 0) tdm_load_ktile_0: ;
  if (wave == 0) {
    tdm_load_2d(Kp, &kstage[0][0], 64, 16, 64);
    __builtin_amdgcn_s_wait_tensorcnt(0);
  }
  if (wave == 1) {
    tdm_load_2d(VT, &vstage[0][0], 32, 64, SS);          // V pair 0
    __builtin_amdgcn_s_wait_tensorcnt(0);
  }
  __syncthreads();
  for (int kt = 0; kt <= ktMax; ++kt) {       // tile count (ktMax+1) is even
    const int cur = kt & 1;
    if (wave == 0 && kt < ktMax)
      tdm_load_2d(Kp + (size_t)(kt + 1) * 16 * DH, &kstage[cur ^ 1][0], 64, 16, 64);
    if (wave == 1 && cur == 1 && kt + 2 <= ktMax) {      // prefetch V pair p+1
      const int pnext = (kt >> 1) + 1;
      tdm_load_2d(VT + (size_t)pnext * 32, &vstage[pnext & 1][0], 32, 64, SS);
    }
    v8h ph;
    if (kt <= ktLim) {
      const _Float16* Krow = &kstage[cur][0];
      v16h ka0 = loadA(Krow, DH, lane);
      v16h ka1 = loadA(Krow + 32, DH, lane);
      v8f st = {};
      st = wmma_f16(ka0, qb0, st);
      st = wmma_f16(ka1, qb1, st);
      v4f p0, p1;
#pragma unroll
      for (int r = 0; r < 8; ++r) {
        int key = kt * 16 + 8 * hi + r;
        float t = st[r] * scale;
        t = (key > qrow) ? -1.0e9f : t;             // masked -> exp underflows
        float p = __expf(t - rmax) * rinv;
        if (r < 4) p0[r] = p; else p1[r - 4] = p;
        ph[r] = (_Float16)p;
      }
      float* dst = P + (size_t)nloc * SS + kt * 16 + 8 * hi;  // 8 consecutive
      *(v4f*)(dst) = p0;
      *(v4f*)(dst + 4) = p1;
    } else {
#pragma unroll
      for (int r = 0; r < 8; ++r) ph[r] = (_Float16)0.0f;
      const v4f z = {0.0f, 0.0f, 0.0f, 0.0f};
      float* dst = P + (size_t)nloc * SS + kt * 16 + 8 * hi;
      *(v4f*)(dst) = z;
      *(v4f*)(dst + 4) = z;
    }
    // one b128 store: P^T row q, keys [(kt&1)*16 + 8*hi, +7]
    *(v8h*)(myp + nloc * 32 + cur * 16 + 8 * hi) = ph;
    if (cur == 1 && (kt & ~1) <= ktLim) {     // pair complete & causally live
      // DS ops are in-order per wave; wait out the return hazard before the
      // cross-lane fragment read (also a compiler memory barrier).
      asm volatile("s_wait_dscnt 0" ::: "memory");
      v16h pf = loadA(myp, 32, lane);         // P as 16x32 A-fragment
      const _Float16* Vtile = &vstage[(kt >> 1) & 1][0];
#pragma unroll
      for (int dt = 0; dt < 4; ++dt) {
        v16h vf = loadB(Vtile + dt * 16 * 32, 32, lane);
        cacc[dt] = wmma_f16(pf, vf, cacc[dt]);
      }
    }
    if (wave <= 1) __builtin_amdgcn_s_wait_tensorcnt(0);
    __syncthreads();
  }

  // context -> [b, s, h*64+d] (f16): rows of the final projection GEMM
  _Float16* C = ctx16 + ((size_t)(b * SS + mbase)) * DD + h * DH;
#pragma unroll
  for (int dt = 0; dt < 4; ++dt)
#pragma unroll
    for (int r = 0; r < 8; ++r)
      C[(size_t)(r + 8 * hi) * DD + dt * 16 + nloc] = (_Float16)cacc[dt][r];

  // zero-fill attn_prob beyond the workgroup frontier (d_out is poisoned)
  const v4f z = {0.0f, 0.0f, 0.0f, 0.0f};
  for (int kt = ktMax + 1; kt < SS / 16; ++kt) {
    float* dst = P + (size_t)nloc * SS + kt * 16 + 8 * hi;
    *(v4f*)(dst) = z;
    *(v4f*)(dst + 4) = z;
  }
}

// ---------------------------------------------------------------- launch
extern "C" void kernel_launch(void* const* d_in, const int* in_sizes, int n_in,
                              void* d_out, int out_size, void* d_ws,
                              size_t ws_size, hipStream_t stream) {
  (void)in_sizes; (void)n_in; (void)out_size; (void)ws_size;
  const float* x  = (const float*)d_in[0];
  // d_in[1] attn_mask: causal triu(k=1); applied analytically (deterministic).
  const float* Wq = (const float*)d_in[2];
  const float* bq = (const float*)d_in[3];
  const float* Wk = (const float*)d_in[4];
  const float* bk = (const float*)d_in[5];
  const float* Wv = (const float*)d_in[6];
  const float* bv = (const float*)d_in[7];
  const float* Wo = (const float*)d_in[8];
  const float* bo = (const float*)d_in[9];

  char* ws = (char*)d_ws;
  _Float16* x16   = (_Float16*)(ws);                 // 16 MB
  _Float16* wqkvT = (_Float16*)(ws + 16777216);      //  6 MB
  _Float16* woT   = (_Float16*)(ws + 23068672);      //  2 MB
  _Float16* q16   = (_Float16*)(ws + 25165824);      // 16 MB
  _Float16* k16   = (_Float16*)(ws + 41943040);      // 16 MB
  _Float16* vT16  = (_Float16*)(ws + 58720256);      // 16 MB
  _Float16* ctx16 = (_Float16*)(ws + 75497472);      // 16 MB (total 88 MB)

  float* outF = (float*)d_out;                       // [4,2048,1024]
  float* prob = outF + (size_t)BB * SS * DD;         // [4,16,2048,2048]

  f32_to_f16_kernel<<<(BB * SS * DD) / 256, 256, 0, stream>>>(x, x16, BB * SS * DD);
  prep_weights_kernel<<<(4 * 1024 * 1024) / 256, 256, 0, stream>>>(
      Wq, Wk, Wv, Wo, wqkvT, woT);
  gemm_kernel<1><<<dim3(3072 / 128, (BB * SS) / 64), 128, 0, stream>>>(
      x16, wqkvT, bq, bk, bv, nullptr, q16, k16, vT16);
  attn_kernel<<<dim3(SS / 64, BB * NH), 128, 0, stream>>>(q16, k16, vT16, ctx16, prob);
  gemm_kernel<0><<<dim3(DD / 128, (BB * SS) / 64), 128, 0, stream>>>(
      ctx16, woT, bo, nullptr, nullptr, outF, nullptr, nullptr, nullptr);
}
